// UberCRSN_1563368096468
// MI455X (gfx1250) — compile-verified, matrix operand order checked by reading, not run on hardware
//
#include <hip/hip_runtime.h>
#include <hip/hip_bf16.h>
#include <math.h>

#define BATCHN  2048
#define DIMN    1024
#define D2N     2048
#define VOCABN  32000
#define NSYMN   64
#define DEPTHN  8
#define STACKN  16
#define EPSF    1e-6f
#define ACTTHF  0.9999f
#define GBSF    0.5f
#define CCF     0.25f

typedef __attribute__((ext_vector_type(16))) __bf16 v16bf;
typedef __attribute__((ext_vector_type(8)))  __bf16 v8bf;
typedef __attribute__((ext_vector_type(8)))  float  v8f;

// ---------------- helpers ----------------

__device__ __forceinline__ unsigned bfbits(float f) {
    unsigned u = __float_as_uint(f);
    return (u + 0x7FFFu + ((u >> 16) & 1u)) >> 16;   // RNE
}

__device__ __forceinline__ __bf16 f2bf(float f) {
    unsigned r = bfbits(f);
    unsigned short s = (unsigned short)r;
    __bf16 h;
    __builtin_memcpy(&h, &s, 2);
    return h;
}

// 16x32 bf16 WMMA A/B fragment from a row-major bf16 matrix: two b128 loads, no VALU.
// Per ISA 7.12.2 (16-bit A-matrix 16x32):
//   lane&15 selects row; lane>>4 selects K-octet: v0..v3 = K kh+[0..7], v4..v7 = K kh+16+[0..7]
__device__ __forceinline__ v16bf load_frag_bf(const __bf16* __restrict__ base, int ld,
                                              int row0, int k0) {
    int lane = threadIdx.x & 31;
    int m  = lane & 15;
    int kh = (lane >> 4) * 8;
    const __bf16* p = base + (size_t)(row0 + m) * ld + k0 + kh;   // 16B aligned
    v8bf lo = *reinterpret_cast<const v8bf*>(p);
    v8bf hi = *reinterpret_cast<const v8bf*>(p + 16);
    return __builtin_shufflevector(lo, hi, 0, 1, 2, 3, 4, 5, 6, 7,
                                           8, 9, 10, 11, 12, 13, 14, 15);
}

__device__ __forceinline__ v8f wmma_bf16(v16bf a, v16bf b, v8f c) {
    return __builtin_amdgcn_wmma_f32_16x16x32_bf16(false, a, false, b,
                                                   (short)0, c, false, false);
}

// C/D 16x16 f32: lane&15 = N, M = r + 8*(lane>>4)
__device__ __forceinline__ void store_tile(float* __restrict__ C, int ldc,
                                           int m0, int n0, v8f acc) {
    int lane = threadIdx.x & 31;
    int n  = n0 + (lane & 15);
    int mb = m0 + ((lane >> 4) * 8);
#pragma unroll
    for (int r = 0; r < 8; ++r) C[(size_t)(mb + r) * ldc + n] = acc[r];
}

__device__ __forceinline__ float blockReduceSum256(float v, float* red) {
    int t = threadIdx.x;
    red[t] = v;
    __syncthreads();
#pragma unroll
    for (int s = 128; s > 0; s >>= 1) {
        if (t < s) red[t] += red[t + s];
        __syncthreads();
    }
    float r = red[0];
    __syncthreads();
    return r;
}

// ---------------- kernels ----------------

// f32 -> bf16 bulk cast: float4 in, packed uint2 (4 x bf16) out.
__global__ void k_cast(const float* __restrict__ src, __bf16* __restrict__ dst, long n4) {
    long i = (long)blockIdx.x * blockDim.x + threadIdx.x;
    if (i >= n4) return;
    float4 v = reinterpret_cast<const float4*>(src)[i];
    unsigned lo = bfbits(v.x) | (bfbits(v.y) << 16);
    unsigned hi = bfbits(v.z) | (bfbits(v.w) << 16);
    reinterpret_cast<uint2*>(dst)[i] = make_uint2(lo, hi);
}

__global__ void k_init(const int* __restrict__ input_ids,
                       const float* __restrict__ emb_mag,
                       const float* __restrict__ emb_phase,
                       float* __restrict__ zr, float* __restrict__ zi,
                       __bf16* __restrict__ zr_bf, __bf16* __restrict__ zi_bf,
                       float* __restrict__ zwr, float* __restrict__ zwi,
                       float* __restrict__ stack_mem, float* __restrict__ stack_ptr,
                       float* __restrict__ halting, float* __restrict__ remain,
                       float* __restrict__ scalars, float* __restrict__ symcnt,
                       float* __restrict__ hist) {
    int b = blockIdx.x, t = threadIdx.x;
    int idx = input_ids[b];
    for (int d = t; d < DIMN; d += 256) {
        float r  = emb_mag[(size_t)idx * DIMN + d];
        float th = emb_phase[(size_t)idx * DIMN + d];
        float a = r * cosf(th);
        float c = r * sinf(th);
        zr[(size_t)b * DIMN + d] = a;
        zi[(size_t)b * DIMN + d] = c;
        zr_bf[(size_t)b * DIMN + d] = f2bf(a);
        zi_bf[(size_t)b * DIMN + d] = f2bf(c);
        zwr[(size_t)b * DIMN + d] = 0.0f;
        zwi[(size_t)b * DIMN + d] = 0.0f;
    }
    size_t smb = (size_t)b * STACKN * D2N;
    for (int c = t; c < STACKN * D2N; c += 256) stack_mem[smb + c] = 0.0f;
    if (t < STACKN) stack_ptr[b * STACKN + t] = (t == 0) ? 1.0f : 0.0f;
    if (t == 0) { halting[b] = 0.0f; remain[b] = 1.0f; }
    if (b == 0) {
        if (t < 4)      scalars[t] = 0.0f;
        if (t < NSYMN)  symcnt[t]  = 0.0f;
        if (t < DEPTHN) hist[t]    = 0.0f;
    }
}

__global__ void k_cbsq(const float* __restrict__ codebook, float* __restrict__ cb_sq) {
    __shared__ float red[256];
    int j = blockIdx.x, t = threadIdx.x;
    float s = 0.0f;
    for (int k = t; k < D2N; k += 256) {
        float v = codebook[(size_t)j * D2N + k];
        s += v * v;
    }
    s = blockReduceSum256(s, red);
    if (t == 0) cb_sq[j] = s;
}

// lr = zr@Wr.T - zi@Wi.T ; li = zi@Wr.T + zr@Wi.T
// Wave tile: 16(M) x 32(N) -> 8 WMMAs per 6 fragment loads, all-bf16 operands.
__global__ void k_cgemm(const __bf16* __restrict__ zr_bf, const __bf16* __restrict__ zi_bf,
                        const __bf16* __restrict__ wr_bf, const __bf16* __restrict__ wi_bf,
                        float* __restrict__ lr, float* __restrict__ li) {
    int wave = (int)((blockIdx.x * blockDim.x + threadIdx.x) >> 5);
    const int NT = DIMN / 32;               // 32 macro-tiles of 2x16 columns
    int m0 = (wave / NT) * 16;
    int n0 = (wave % NT) * 32;
    v8f aRR0 = {}, aRR1 = {}, aII0 = {}, aII1 = {};
    v8f aIR0 = {}, aIR1 = {}, aRI0 = {}, aRI1 = {};
    for (int k0 = 0; k0 < DIMN; k0 += 32) {
        v16bf ar  = load_frag_bf(zr_bf, DIMN, m0, k0);
        v16bf ai  = load_frag_bf(zi_bf, DIMN, m0, k0);
        v16bf br0 = load_frag_bf(wr_bf, DIMN, n0,      k0);   // B[k][n] = Wr[n][k]
        v16bf br1 = load_frag_bf(wr_bf, DIMN, n0 + 16, k0);
        v16bf bi0 = load_frag_bf(wi_bf, DIMN, n0,      k0);
        v16bf bi1 = load_frag_bf(wi_bf, DIMN, n0 + 16, k0);
        aRR0 = wmma_bf16(ar, br0, aRR0);
        aRR1 = wmma_bf16(ar, br1, aRR1);
        aII0 = wmma_bf16(ai, bi0, aII0);
        aII1 = wmma_bf16(ai, bi1, aII1);
        aIR0 = wmma_bf16(ai, br0, aIR0);
        aIR1 = wmma_bf16(ai, br1, aIR1);
        aRI0 = wmma_bf16(ar, bi0, aRI0);
        aRI1 = wmma_bf16(ar, bi1, aRI1);
    }
    store_tile(lr, DIMN, m0, n0,      aRR0 - aII0);
    store_tile(lr, DIMN, m0, n0 + 16, aRR1 - aII1);
    store_tile(li, DIMN, m0, n0,      aIR0 + aRI0);
    store_tile(li, DIMN, m0, n0 + 16, aIR1 + aRI1);
}

// One block per batch row: mag-LN + modReLU + halt/stack heads + soft stack + read.
__global__ void k_rownorm(const float* __restrict__ lr, const float* __restrict__ li,
                          const float* __restrict__ ln_scale, const float* __restrict__ ln_shift,
                          const float* __restrict__ mr_bias,
                          const float* __restrict__ halt_W, const float* __restrict__ halt_b,
                          const float* __restrict__ stk_W, const float* __restrict__ stk_b,
                          float* __restrict__ zcf,
                          float* __restrict__ stack_mem, float* __restrict__ stack_ptr,
                          float* __restrict__ p_halt_arr, float* __restrict__ hist, int step) {
    __shared__ float red[256];
    __shared__ float sp[STACKN], npv[STACKN], wmv[STACKN];
    __shared__ float s_ctrl[3];
    __shared__ float s_inv;
    int b = blockIdx.x, t = threadIdx.x;
    const float* Lr = lr + (size_t)b * DIMN;
    const float* Li = li + (size_t)b * DIMN;

    float ar[4], ai[4], mg[4];
    float s1 = 0.0f, s2 = 0.0f;
#pragma unroll
    for (int j = 0; j < 4; ++j) {
        int d = t + 256 * j;
        float x = Lr[d], y = Li[d];
        float h = sqrtf(x * x + y * y);
        float m = h + EPSF;
        ar[j] = x; ai[j] = y; mg[j] = m;
        s1 += m; s2 += m * m;
    }
    s1 = blockReduceSum256(s1, red);
    s2 = blockReduceSum256(s2, red);
    float mean   = s1 / (float)DIMN;
    float var    = (s2 - (float)DIMN * mean * mean) / (float)(DIMN - 1); // ddof=1
    float invstd = rsqrtf(var + EPSF);

    float pr[4], pi[4];
    float dh = 0.0f, d0 = 0.0f, d1 = 0.0f, d2 = 0.0f;
#pragma unroll
    for (int j = 0; j < 4; ++j) {
        int d = t + 256 * j;
        float nm   = (mg[j] - mean) * invstd * ln_scale[d] + ln_shift[d];
        float hyp  = mg[j] - EPSF;
        float invh = 1.0f / fmaxf(hyp, 1e-30f);
        float cph = ar[j] * invh, sph = ai[j] * invh;       // cos/sin of atan2(li,lr)
        float nr = nm * cph, ni = nm * sph;
        float nrm = sqrtf(nr * nr + ni * ni) + EPSF;
        float sc  = fmaxf(nrm + mr_bias[d], 0.0f) / nrm;    // modReLU
        float prv = nr * sc, piv = ni * sc;
        pr[j] = prv; pi[j] = piv;
        dh += prv * halt_W[d] + piv * halt_W[DIMN + d];
        d0 += prv * stk_W[0 * D2N + d] + piv * stk_W[0 * D2N + DIMN + d];
        d1 += prv * stk_W[1 * D2N + d] + piv * stk_W[1 * D2N + DIMN + d];
        d2 += prv * stk_W[2 * D2N + d] + piv * stk_W[2 * D2N + DIMN + d];
    }
    dh = blockReduceSum256(dh, red);
    d0 = blockReduceSum256(d0, red);
    d1 = blockReduceSum256(d1, red);
    d2 = blockReduceSum256(d2, red);
    if (t == 0) {
        p_halt_arr[b] = 1.0f / (1.0f + expf(-(dh + halt_b[0])));
        float l0 = d0 + stk_b[0], l1 = d1 + stk_b[1], l2 = d2 + stk_b[2];
        float mx = fmaxf(l0, fmaxf(l1, l2));
        float e0 = expf(l0 - mx), e1 = expf(l1 - mx), e2 = expf(l2 - mx);
        float inv = 1.0f / (e0 + e1 + e2);
        s_ctrl[0] = e0 * inv; s_ctrl[1] = e1 * inv; s_ctrl[2] = e2 * inv;
    }
    if (t < STACKN) sp[t] = stack_ptr[b * STACKN + t];
    __syncthreads();
    if (t < STACKN) {
        float push = s_ctrl[0], pop = s_ctrl[1], noop = s_ctrl[2];
        float up = sp[(t + STACKN - 1) & (STACKN - 1)];   // roll(+1)
        float dn = sp[(t + 1) & (STACKN - 1)];            // roll(-1)
        npv[t] = push * up + pop * dn + noop * sp[t];
        wmv[t] = push * up;
    }
    __syncthreads();
    if (t == 0) {
        float s = 0.0f;
        for (int i = 0; i < STACKN; ++i) s += npv[i];
        s_inv = 1.0f / (s + EPSF);
    }
    __syncthreads();
    if (t < STACKN) {
        npv[t] *= s_inv;
        stack_ptr[b * STACKN + t] = npv[t];
    }
    __syncthreads();
    if (t == 0) {
        int am = 0; float bv = npv[0];
        for (int i = 1; i < STACKN; ++i) if (npv[i] > bv) { bv = npv[i]; am = i; }
        atomicAdd(&hist[step], (float)am / (float)BATCHN);
    }

    // Soft stack write + weighted read: single pass over B x 16 x 2048.
    size_t smb = (size_t)b * STACKN * D2N;
#pragma unroll
    for (int jj = 0; jj < 8; ++jj) {
        int c = t + 256 * jj;                      // column in [0, 2048)
        float zv = (jj < 4) ? pr[jj] : pi[jj - 4]; // zf[b][c]
        float racc = 0.0f;
#pragma unroll
        for (int s = 0; s < STACKN; ++s) {
            size_t p = smb + (size_t)s * D2N + c;
            float v = stack_mem[p];
            v = wmv[s] * zv + v * (1.0f - wmv[s]);
            stack_mem[p] = v;
            racc += v * npv[s];
        }
        zcf[(size_t)b * D2N + c] = zv + racc;      // sr = pr+read, si = pi+read
    }
}

// VQ + straight-through mix + ACT halting update (forward: stop_gradient == identity).
__global__ void k_vq(const float* __restrict__ zcf, const float* __restrict__ codebook,
                     const float* __restrict__ cb_sq, const float* __restrict__ adjacency,
                     int* __restrict__ cur_sym,
                     float* __restrict__ zr, float* __restrict__ zi,
                     __bf16* __restrict__ zr_bf, __bf16* __restrict__ zi_bf,
                     float* __restrict__ zwr, float* __restrict__ zwi,
                     float* __restrict__ halting, float* __restrict__ remain,
                     const float* __restrict__ p_halt_arr,
                     float* __restrict__ scalars, float* __restrict__ symcnt, int step) {
    __shared__ float s_z[D2N];
    __shared__ float red[256];
    __shared__ float s_d[NSYMN];
    __shared__ float sh[2];   // [0]=|z|^2, [1]=p
    __shared__ int   s_mi;
    int b = blockIdx.x, t = threadIdx.x;
    const float* Z = zcf + (size_t)b * D2N;

    // Stage the 8KB zcf row straight into LDS via the CDNA5 async copy path
    // (GLOBAL_LOAD_ASYNC_TO_LDS_B128, tracked by ASYNCcnt): 2 x 16B per thread,
    // no VGPR round-trip. INST_OFFSET applies to both LDS and global addresses.
    {
        int c0 = t * 8;                                    // 8 floats = 32B per thread
        unsigned  laddr = (unsigned)(size_t)(&s_z[c0]);    // LDS byte offset (addr[31:0])
        const float* g  = Z + c0;
        asm volatile("global_load_async_to_lds_b128 %0, %1, off"
                     :: "v"(laddr), "v"(g) : "memory");
        asm volatile("global_load_async_to_lds_b128 %0, %1, off offset:16"
                     :: "v"(laddr), "v"(g) : "memory");
        asm volatile("s_wait_asynccnt 0" ::: "memory");
    }
    __syncthreads();

    float zs = 0.0f;
#pragma unroll
    for (int jj = 0; jj < 8; ++jj) {
        float v = s_z[t + 256 * jj];
        zs += v * v;
    }
    zs = blockReduceSum256(zs, red);
    if (t == 0) sh[0] = zs;
    int prev = cur_sym[b];     // read before the write below (use guarded by step>0)
    __syncthreads();
    if (t < NSYMN) {
        const float* cb = codebook + (size_t)t * D2N;
        float dot = 0.0f;
        for (int k = 0; k < D2N; ++k) dot += s_z[k] * cb[k];
        float dj = sh[0] + cb_sq[t] - 2.0f * dot;
        if (step > 0) {
            float a = adjacency[(size_t)prev * NSYMN + t];
            dj -= GBSF * (1.0f / (1.0f + expf(-a)));
        }
        s_d[t] = dj;
    }
    __syncthreads();
    if (t == 0) {
        int mi = 0; float bv = s_d[0];
        for (int i = 1; i < NSYMN; ++i) if (s_d[i] < bv) { bv = s_d[i]; mi = i; }
        s_mi = mi;
        cur_sym[b] = mi;
        atomicAdd(&symcnt[mi], 1.0f);
        float h = halting[b];
        float still = (h < ACTTHF) ? 1.0f : 0.0f;
        float p = (step == DEPTHN - 1) ? remain[b] : p_halt_arr[b] * still;
        sh[1] = p;
        halting[b] = h + p;
        remain[b]  = remain[b] - p;
        atomicAdd(&scalars[0], still / (float)BATCHN);      // ponder
    }
    __syncthreads();
    int mi = s_mi;
    float p = sh[1];
    const float* q = codebook + (size_t)mi * D2N;
    float vqs = 0.0f;
#pragma unroll
    for (int jj = 0; jj < 8; ++jj) {
        int c = t + 256 * jj;
        float zc = s_z[c], zq = q[c];
        float df = zq - zc;
        vqs += df * df;
        float zn = 0.7f * zc + 0.3f * zq;   // zq_st == zq in forward
        if (c < DIMN) {
            zr[(size_t)b * DIMN + c] = zn;
            zr_bf[(size_t)b * DIMN + c] = f2bf(zn);
            zwr[(size_t)b * DIMN + c] += p * zn;
        } else {
            int d = c - DIMN;
            zi[(size_t)b * DIMN + d] = zn;
            zi_bf[(size_t)b * DIMN + d] = f2bf(zn);
            zwi[(size_t)b * DIMN + d] += p * zn;
        }
    }
    vqs = blockReduceSum256(vqs, red);
    if (t == 0)
        atomicAdd(&scalars[1], (1.0f + CCF) * vqs / ((float)BATCHN * 2.0f * DIMN));
}

__global__ void k_perp(float* __restrict__ symcnt, float* __restrict__ scalars) {
    __shared__ float red[NSYMN];
    int t = threadIdx.x;
    float cnt = symcnt[t];
    symcnt[t] = 0.0f;                              // reset for next step
    float avg = cnt / (float)BATCHN;
    red[t] = -avg * logf(avg + 1e-10f);
    __syncthreads();
#pragma unroll
    for (int s = 32; s > 0; s >>= 1) {
        if (t < s) red[t] += red[t + s];
        __syncthreads();
    }
    if (t == 0) scalars[2] += expf(red[0]);
}

// logits = [zwr|zwi] @ dec_W.T + dec_b, all-bf16 operands.
// Wave tile = 64(M) x 16(N): B-fragment reused across 4 M-tiles; dec_W(bf16, 131MB)
// is fully L2-resident on MI455X (192MB L2).
__global__ void k_dec(const __bf16* __restrict__ zwr_bf, const __bf16* __restrict__ zwi_bf,
                      const __bf16* __restrict__ dec_bf, const float* __restrict__ dec_b,
                      float* __restrict__ logits) {
    int wave = (int)((blockIdx.x * blockDim.x + threadIdx.x) >> 5);
    const int MT = BATCHN / 64;                    // 32 M macro-tiles
    int n0 = (wave / MT) * 16;                     // consecutive waves share n0
    int m0 = (wave % MT) * 64;
    v8f a0 = {}, a1 = {}, a2 = {}, a3 = {};
    for (int k0 = 0; k0 < D2N; k0 += 32) {
        const __bf16* A = (k0 < DIMN) ? zwr_bf : zwi_bf;   // features = concat(zwr, zwi)
        int ka = (k0 < DIMN) ? k0 : (k0 - DIMN);
        if (k0 + 32 < D2N)
            __builtin_prefetch(dec_bf + (size_t)(n0 + (threadIdx.x & 15)) * D2N + k0 + 32, 0, 1);
        v16bf f0 = load_frag_bf(A, DIMN, m0,      ka);
        v16bf f1 = load_frag_bf(A, DIMN, m0 + 16, ka);
        v16bf f2 = load_frag_bf(A, DIMN, m0 + 32, ka);
        v16bf f3 = load_frag_bf(A, DIMN, m0 + 48, ka);
        v16bf bw = load_frag_bf(dec_bf, D2N, n0, k0);
        a0 = wmma_bf16(f0, bw, a0);
        a1 = wmma_bf16(f1, bw, a1);
        a2 = wmma_bf16(f2, bw, a2);
        a3 = wmma_bf16(f3, bw, a3);
    }
    int lane = threadIdx.x & 31;
    int n = n0 + (lane & 15);
    float bias = dec_b[n];
    v8f acc[4] = { a0, a1, a2, a3 };
#pragma unroll
    for (int u = 0; u < 4; ++u) {
        int mb = m0 + u * 16 + ((lane >> 4) * 8);
#pragma unroll
        for (int r = 0; r < 8; ++r)
            logits[(size_t)(mb + r) * VOCABN + n] = acc[u][r] + bias;
    }
}

__global__ void k_final(const float* __restrict__ zwr, const float* __restrict__ zwi,
                        const int* __restrict__ cur_sym,
                        const float* __restrict__ scalars, const float* __restrict__ hist,
                        float* __restrict__ o_zwr, float* __restrict__ o_zwi,
                        float* __restrict__ o_sym, float* __restrict__ o_sc,
                        float* __restrict__ o_hist) {
    size_t i = (size_t)blockIdx.x * blockDim.x + threadIdx.x;
    size_t ND = (size_t)BATCHN * DIMN;
    if (i < ND) { o_zwr[i] = zwr[i]; o_zwi[i] = zwi[i]; }
    if (i < BATCHN) o_sym[i] = (float)cur_sym[i];
    if (i == 0) {
        o_sc[0] = scalars[0];                    // ponder
        o_sc[1] = scalars[1];                    // vq_total
        o_sc[2] = scalars[2] / (float)DEPTHN;    // perp_total / DEPTH
    }
    if (i < DEPTHN) o_hist[i] = hist[i];
}

// ---------------- host ----------------

extern "C" void kernel_launch(void* const* d_in, const int* in_sizes, int n_in,
                              void* d_out, int out_size, void* d_ws, size_t ws_size,
                              hipStream_t stream) {
    (void)in_sizes; (void)n_in; (void)out_size; (void)ws_size;
    const int*   input_ids = (const int*)  d_in[0];
    const float* emb_mag   = (const float*)d_in[1];
    const float* emb_phase = (const float*)d_in[2];
    const float* Wr        = (const float*)d_in[3];
    const float* Wi        = (const float*)d_in[4];
    const float* ln_scale  = (const float*)d_in[5];
    const float* ln_shift  = (const float*)d_in[6];
    const float* mr_bias   = (const float*)d_in[7];
    const float* halt_W    = (const float*)d_in[8];
    const float* halt_b    = (const float*)d_in[9];
    const float* stk_W     = (const float*)d_in[10];
    const float* stk_b     = (const float*)d_in[11];
    const float* codebook  = (const float*)d_in[12];
    const float* adjacency = (const float*)d_in[13];
    const float* dec_W     = (const float*)d_in[14];
    const float* dec_b     = (const float*)d_in[15];

    float* W = (float*)d_ws;
    size_t off = 0;
    auto alloc = [&](size_t n) { float* p = W + off; off += n; return p; };
    // All region sizes are multiples of 4 floats -> every region is 16B aligned.
    float* zr        = alloc((size_t)BATCHN * DIMN);
    float* zi        = alloc((size_t)BATCHN * DIMN);
    float* lr        = alloc((size_t)BATCHN * DIMN);
    float* li        = alloc((size_t)BATCHN * DIMN);
    float* zcf       = alloc((size_t)BATCHN * D2N);
    float* stack_mem = alloc((size_t)BATCHN * STACKN * D2N);
    float* stack_ptr = alloc((size_t)BATCHN * STACKN);
    float* halting   = alloc(BATCHN);
    float* remain    = alloc(BATCHN);
    float* p_halt    = alloc(BATCHN);
    float* zwr       = alloc((size_t)BATCHN * DIMN);
    float* zwi       = alloc((size_t)BATCHN * DIMN);
    float* cb_sq     = alloc(NSYMN);
    float* symcnt    = alloc(NSYMN);
    float* hist      = alloc(DEPTHN);
    float* scalars   = alloc(4);      // [0]=ponder [1]=vq_total [2]=perp_total
    int*   cur_sym   = (int*)alloc(BATCHN);
    // bf16 regions (2 floats of ws per 4 bf16 elements)
    __bf16* wr_bf  = (__bf16*)alloc((size_t)DIMN * DIMN / 2);
    __bf16* wi_bf  = (__bf16*)alloc((size_t)DIMN * DIMN / 2);
    __bf16* dec_bf = (__bf16*)alloc((size_t)VOCABN * D2N / 2);
    __bf16* zr_bf  = (__bf16*)alloc((size_t)BATCHN * DIMN / 2);
    __bf16* zi_bf  = (__bf16*)alloc((size_t)BATCHN * DIMN / 2);
    __bf16* zwr_bf = (__bf16*)alloc((size_t)BATCHN * DIMN / 2);
    __bf16* zwi_bf = (__bf16*)alloc((size_t)BATCHN * DIMN / 2);

    float* logits = (float*)d_out;
    float* o_zwr  = logits + (size_t)BATCHN * VOCABN;
    float* o_zwi  = o_zwr + (size_t)BATCHN * DIMN;
    float* o_sym  = o_zwi + (size_t)BATCHN * DIMN;
    float* o_sc   = o_sym + BATCHN;
    float* o_hist = o_sc + 3;

    // One-time weight casts (dec_W bf16 = 131MB -> L2-resident for the decoder GEMM).
    {
        long n4w = (long)DIMN * DIMN / 4;
        k_cast<<<(int)((n4w + 255) / 256), 256, 0, stream>>>(Wr, wr_bf, n4w);
        k_cast<<<(int)((n4w + 255) / 256), 256, 0, stream>>>(Wi, wi_bf, n4w);
        long n4d = (long)VOCABN * D2N / 4;
        k_cast<<<(int)((n4d + 255) / 256), 256, 0, stream>>>(dec_W, dec_bf, n4d);
    }

    k_init<<<BATCHN, 256, 0, stream>>>(input_ids, emb_mag, emb_phase, zr, zi,
                                       zr_bf, zi_bf, zwr, zwi,
                                       stack_mem, stack_ptr, halting, remain,
                                       scalars, symcnt, hist);
    k_cbsq<<<NSYMN, 256, 0, stream>>>(codebook, cb_sq);

    for (int step = 0; step < DEPTHN; ++step) {
        // tiles: (2048/16)*(1024/32) = 4096 waves -> 512 blocks of 8 waves
        k_cgemm<<<512, 256, 0, stream>>>(zr_bf, zi_bf, wr_bf, wi_bf, lr, li);
        k_rownorm<<<BATCHN, 256, 0, stream>>>(lr, li, ln_scale, ln_shift, mr_bias,
                                              halt_W, halt_b, stk_W, stk_b,
                                              zcf, stack_mem, stack_ptr,
                                              p_halt, hist, step);
        k_vq<<<BATCHN, 256, 0, stream>>>(zcf, codebook, cb_sq, adjacency, cur_sym,
                                         zr, zi, zr_bf, zi_bf, zwr, zwi,
                                         halting, remain, p_halt,
                                         scalars, symcnt, step);
        k_perp<<<1, NSYMN, 0, stream>>>(symcnt, scalars);
    }

    // Cast accumulated features once, then decoder GEMM.
    {
        long n4f = (long)BATCHN * DIMN / 4;
        k_cast<<<(int)((n4f + 255) / 256), 256, 0, stream>>>(zwr, zwr_bf, n4f);
        k_cast<<<(int)((n4f + 255) / 256), 256, 0, stream>>>(zwi, zwi_bf, n4f);
    }
    // decoder: (32000/16)*(2048/64) = 64000 waves -> 8000 blocks of 8 waves
    k_dec<<<8000, 256, 0, stream>>>(zwr_bf, zwi_bf, dec_bf, dec_b, logits);
    k_final<<<(BATCHN * DIMN + 255) / 256, 256, 0, stream>>>(zwr, zwi, cur_sym,
                                                             scalars, hist,
                                                             o_zwr, o_zwi, o_sym,
                                                             o_sc, o_hist);
}
